// Masked_Dilated_WSA_50371376447936
// MI455X (gfx1250) — matrix operand order repaired and checked by reading, not compiled, not run on hardware
//
#include <hip/hip_runtime.h>
#include <hip/hip_bf16.h>

typedef _Float16 f16;
typedef __attribute__((ext_vector_type(16))) _Float16 v16h;
typedef __attribute__((ext_vector_type(8)))  _Float16 v8h;
typedef __attribute__((ext_vector_type(8)))  float    v8f;

#define CDIM   192
#define WSZ    12
#define NTOK   144
#define NHEADS 6
#define DH     32
#define IMG    256
#define NWIN   22      // 264/12

// LDS strides (in elements), padded to dodge bank conflicts
#define SX   200       // X / O region   : [144][200] f16
#define SQK  392       // Q|K region     : [144][392] f16  (Q ch 0..191, K ch 192..383)
#define SV   168       // Vt region      : [192][168] f16  ([head*32+d][token], K padded->160)
#define SP   168       // P strips       : 8 waves x [16][168] f16
#define SSTG 148       // proj staging   : [192][148] f32  (aliases QK+Vt)

#define QKV_B (NTOK * SQK * 2)          // 112896
#define VT_B  (CDIM * SV * 2)           //  64512
#define P_B   (8 * 16 * SP * 2)         //  43008

union HF16 { v16h v; v8h h[2]; };

__device__ __forceinline__ v8f vzero8() {
    v8f z;
#pragma unroll
    for (int i = 0; i < 8; ++i) z[i] = 0.0f;
    return z;
}

// A-fragment (16x32 f16), base is [M][K] row-major:
//   lane<16 -> row=lane,    K = kb+0..15
//   lane>=16 -> row=lane-16, K = kb+16..31
__device__ __forceinline__ v16h load_afrag(const f16* base, int stride, int row0,
                                           int kbase, int lane) {
    const f16* p = base + (row0 + (lane & 15)) * stride + kbase + ((lane >> 4) << 4);
    HF16 u;
    u.h[0] = *(const v8h*)(p);
    u.h[1] = *(const v8h*)(p + 8);
    return u.v;
}

// B-fragment (32x16 f16), base is [N][K] row-major:
//   lane holds column N=col0+(lane&15); K contiguous 16:
//   lanes<16 -> kb+0..15, lanes>=16 -> kb+16..31
__device__ __forceinline__ v16h load_bfrag(const f16* base, int stride, int col0,
                                           int kbase, int lane) {
    const f16* p = base + (col0 + (lane & 15)) * stride + kbase + ((lane >> 4) << 4);
    HF16 u;
    u.h[0] = *(const v8h*)(p);
    u.h[1] = *(const v8h*)(p + 8);
    return u.v;
}

__device__ __forceinline__ v8f wmma16(v16h a, v16h b, v8f c) {
    return __builtin_amdgcn_wmma_f32_16x16x32_f16(false, a, false, b, (short)0, c,
                                                  false, false);
}

// ---------------------------------------------------------------- prep kernel
__global__ void wsa_prep_weights(const float* __restrict__ wqkv,
                                 const float* __restrict__ wproj,
                                 f16* __restrict__ wq_h, f16* __restrict__ wp_h) {
    int idx = blockIdx.x * 256 + threadIdx.x;
    const int NQ = 3 * CDIM * CDIM;   // 110592
    const int NP = CDIM * CDIM;       // 36864
    if (idx < NQ)            wq_h[idx] = (f16)wqkv[idx];
    else if (idx < NQ + NP)  wp_h[idx - NQ] = (f16)wproj[idx - NQ];
}

// ---------------------------------------------------------------- main kernel
__global__ __launch_bounds__(256)
void wsa_fused(const float* __restrict__ x, const f16* __restrict__ wqkv_h,
               const f16* __restrict__ wproj_h, float* __restrict__ out) {
    __shared__ __align__(16) f16 sX[NTOK * SX];                       // 57600 B
    __shared__ __align__(16) unsigned char sHeap[QKV_B + VT_B + P_B]; // 220416 B

    f16* sQK = (f16*)sHeap;
    f16* sVt = (f16*)(sHeap + QKV_B);
    f16* sP  = (f16*)(sHeap + QKV_B + VT_B);

    const int tid  = threadIdx.x;
    const int lane = tid & 31;
    const int wv   = tid >> 5;

    // window decode
    int bw  = blockIdx.x;
    int b   = bw / (NWIN * NWIN);
    int r   = bw - b * (NWIN * NWIN);
    int wh0 = (r / NWIN) * WSZ;
    int ww0 = (r % NWIN) * WSZ;
    bool interior_w = (ww0 + WSZ) <= IMG;   // no right-edge reflect needed

    // ---------------- Phase 1: gather window (reflect pad) -> f16 LDS -------
    for (int t = tid; t < CDIM * WSZ; t += 256) {
        int c  = t / WSZ;
        int hl = t - c * WSZ;
        int h  = wh0 + hl;
        int hs = (h < IMG) ? h : (2 * IMG - 2 - h);
        const float* row = x + ((size_t)((b * CDIM + c) * IMG + hs)) * IMG;
        f16* xrow = sX + (hl * WSZ) * SX + c;
        if (interior_w) {
            const float4* p = (const float4*)(row + ww0);   // 48B-aligned
            float4 f0 = p[0], f1 = p[1], f2 = p[2];
            xrow[0 * SX] = (f16)f0.x;  xrow[1 * SX] = (f16)f0.y;
            xrow[2 * SX] = (f16)f0.z;  xrow[3 * SX] = (f16)f0.w;
            xrow[4 * SX] = (f16)f1.x;  xrow[5 * SX] = (f16)f1.y;
            xrow[6 * SX] = (f16)f1.z;  xrow[7 * SX] = (f16)f1.w;
            xrow[8 * SX] = (f16)f2.x;  xrow[9 * SX] = (f16)f2.y;
            xrow[10 * SX] = (f16)f2.z; xrow[11 * SX] = (f16)f2.w;
        } else {
#pragma unroll
            for (int wl = 0; wl < WSZ; ++wl) {
                int wg = ww0 + wl;
                int ws = (wg < IMG) ? wg : (2 * IMG - 2 - wg);
                xrow[wl * SX] = (f16)row[ws];
            }
        }
    }
    // zero Vt K-padding (tokens 144..159)
    for (int t = tid; t < CDIM * 16; t += 256) {
        int cr = t >> 4;
        sVt[cr * SV + NTOK + (t & 15)] = (f16)0.0f;
    }
    __syncthreads();

    // ---------------- Phase 2: QKV GEMM, swapped: (576x192) @ (192x144) -----
    // A = w_qkv rows (M = out-channel, 36 tiles, blocked in pairs),
    // B = X (N = token, 9 tiles). C lane holds one token, 8 consecutive chs.
    {
        const int n  = lane & 15;
        const int mO = (lane >> 4) * 8;
        for (int i = wv; i < 18 * 9; i += 8) {
            int mp  = i / 9;            // M-tile pair 0..17 (never straddles QK|V at 24)
            int tn  = i % 9;
            int tm0 = mp * 2;
            v8f acc0 = vzero8(), acc1 = vzero8();
#pragma unroll
            for (int kc = 0; kc < 6; ++kc) {
                v16h bb = load_bfrag(sX, SX, tn * 16, kc * 32, lane);
                v16h a0 = load_afrag(wqkv_h, CDIM, tm0 * 16, kc * 32, lane);
                v16h a1 = load_afrag(wqkv_h, CDIM, tm0 * 16 + 16, kc * 32, lane);
                acc0 = wmma16(a0, bb, acc0);
                acc1 = wmma16(a1, bb, acc1);
            }
            int token = tn * 16 + n;
#pragma unroll
            for (int s = 0; s < 2; ++s) {
                int tm = tm0 + s;
                v8f& acc = s ? acc1 : acc0;
                if (tm < 24) {                 // Q|K: [token][ch], 8 contiguous chs
                    v8h hv;
#pragma unroll
                    for (int j = 0; j < 8; ++j) hv[j] = (f16)acc[j];
                    *(v8h*)(sQK + token * SQK + tm * 16 + mO) = hv;
                } else {                       // V: transposed [ch][token]
                    int cr0 = (tm - 24) * 16 + mO;
#pragma unroll
                    for (int j = 0; j < 8; ++j)
                        sVt[(cr0 + j) * SV + token] = (f16)acc[j];
                }
            }
        }
    }
    __syncthreads();

    // ---------------- Phase 3: attention (one wave = one head x row-tile) ---
    const float scale = 0.17677669529663687f;   // 32^-0.5
    f16* sPw = sP + wv * 16 * SP;
    {   // zero P K-padding (cols 144..159) once, rows 0..15
        v8h z{};
        int rr = lane & 15;
        int cb = NTOK + ((lane >> 4) << 3);
        *(v8h*)(sPw + rr * SP + cb) = z;
    }

    for (int i = wv; i < NHEADS * 9; i += 8) {
        int head = i / 9;
        int rt   = i % 9;
        int n    = lane & 15;
        int mO   = (lane >> 4) * 8;

        v16h qf = load_afrag(sQK, SQK, rt * 16, head * DH, lane);
        v8f S[9];
#pragma unroll
        for (int nt = 0; nt < 9; ++nt) {
            v16h kf = load_bfrag(sQK, SQK, nt * 16, CDIM + head * DH, lane);
            S[nt] = wmma16(qf, kf, vzero8());
        }

        // scale + dilated mask + row max (rows live in 16-lane halves)
        float rmax[8];
#pragma unroll
        for (int j = 0; j < 8; ++j) rmax[j] = -1e30f;
#pragma unroll
        for (int nt = 0; nt < 9; ++nt) {
            int ki = nt * 16 + n;
            int kh = ki / WSZ, kw = ki - kh * WSZ;
            int kh3 = kh % 3, kw3 = kw % 3;
#pragma unroll
            for (int j = 0; j < 8; ++j) {
                int qi = rt * 16 + j + mO;
                int qh = qi / WSZ, qw = qi - qh * WSZ;
                bool ok = ((qh % 3) == kh3) && ((qw % 3) == kw3) && (qi != ki);
                float s = ok ? (S[nt][j] * scale) : -1e30f;
                S[nt][j] = s;
                rmax[j] = fmaxf(rmax[j], s);
            }
        }
#pragma unroll
        for (int j = 0; j < 8; ++j) {
#pragma unroll
            for (int m = 1; m < 16; m <<= 1)
                rmax[j] = fmaxf(rmax[j], __shfl_xor(rmax[j], m, 32));
        }

        // exp, row sum, spill unnormalized P (f16) to wave-private strip
        float rsum[8];
#pragma unroll
        for (int j = 0; j < 8; ++j) rsum[j] = 0.0f;
#pragma unroll
        for (int nt = 0; nt < 9; ++nt) {
#pragma unroll
            for (int j = 0; j < 8; ++j) {
                float e = __expf(S[nt][j] - rmax[j]);
                rsum[j] += e;
                sPw[(j + mO) * SP + nt * 16 + n] = (f16)e;
            }
        }
#pragma unroll
        for (int j = 0; j < 8; ++j) {
#pragma unroll
            for (int m = 1; m < 16; m <<= 1)
                rsum[j] += __shfl_xor(rsum[j], m, 32);
        }

        asm volatile("s_wait_dscnt 0x0" ::: "memory");  // P stores visible to us

        // O = P @ Vt   (K padded to 160 -> 5 chunks of 32)
        v8f O0 = vzero8(), O1 = vzero8();
        const f16* vb = sVt + head * DH * SV;
#pragma unroll
        for (int kc = 0; kc < 5; ++kc) {
            v16h pf = load_afrag(sPw, SP, 0, kc * 32, lane);
            O0 = wmma16(pf, load_bfrag(vb, SV, 0,  kc * 32, lane), O0);
            O1 = wmma16(pf, load_bfrag(vb, SV, 16, kc * 32, lane), O1);
        }
#pragma unroll
        for (int j = 0; j < 8; ++j) {
            float inv = 1.0f / rsum[j];
            f16* orow = sX + (rt * 16 + j + mO) * SX + head * DH;
            orow[n]      = (f16)(O0[j] * inv);
            orow[16 + n] = (f16)(O1[j] * inv);
        }
    }
    __syncthreads();

    // ---------------- Phase 4: projection (144x192 @ 192x192), N-pairs ------
    float* sStg = (float*)sHeap;   // [192][148] f32, aliases dead QK+Vt
    for (int i = wv; i < 9 * 6; i += 8) {
        int tm  = i / 6;
        int tn0 = (i % 6) * 2;
        v8f acc0 = vzero8(), acc1 = vzero8();
#pragma unroll
        for (int kc = 0; kc < 6; ++kc) {
            v16h a  = load_afrag(sX, SX, tm * 16, kc * 32, lane);
            v16h b0 = load_bfrag(wproj_h, CDIM, tn0 * 16, kc * 32, lane);
            v16h b1 = load_bfrag(wproj_h, CDIM, tn0 * 16 + 16, kc * 32, lane);
            acc0 = wmma16(a, b0, acc0);
            acc1 = wmma16(a, b1, acc1);
        }
        int tok0 = tm * 16 + (lane >> 4) * 8;
#pragma unroll
        for (int s = 0; s < 2; ++s) {
            v8f& acc = s ? acc1 : acc0;
            int ch = (tn0 + s) * 16 + (lane & 15);
            float* p = sStg + ch * SSTG + tok0;
            *(float4*)(p)     = make_float4(acc[0], acc[1], acc[2], acc[3]);
            *(float4*)(p + 4) = make_float4(acc[4], acc[5], acc[6], acc[7]);
        }
    }
    __syncthreads();

    // ---------------- Phase 5: cropped NCHW store ---------------------------
    for (int t = tid; t < CDIM * WSZ; t += 256) {
        int c  = t / WSZ;
        int hl = t - c * WSZ;
        int h  = wh0 + hl;
        if (h < IMG) {
            float* orow = out + ((size_t)((b * CDIM + c) * IMG + h)) * IMG;
            const float* srow = sStg + c * SSTG + hl * WSZ;
#pragma unroll
            for (int wl = 0; wl < WSZ; ++wl) {
                int w = ww0 + wl;
                if (w < IMG) orow[w] = srow[wl];
            }
        }
    }
}

// ---------------------------------------------------------------- launcher
extern "C" void kernel_launch(void* const* d_in, const int* in_sizes, int n_in,
                              void* d_out, int out_size, void* d_ws, size_t ws_size,
                              hipStream_t stream) {
    const float* x     = (const float*)d_in[0];
    const float* wqkv  = (const float*)d_in[1];
    const float* wproj = (const float*)d_in[2];

    f16* wq_h = (f16*)d_ws;                       // 110592 f16
    f16* wp_h = wq_h + 3 * CDIM * CDIM;           //  36864 f16

    wsa_prep_weights<<<576, 256, 0, stream>>>(wqkv, wproj, wq_h, wp_h);

    int nwin = 4 * NWIN * NWIN;                   // 1936 windows
    wsa_fused<<<nwin, 256, 0, stream>>>(x, wq_h, wp_h, (float*)d_out);
}
